// MLLABlock_9242769620997
// MI455X (gfx1250) — matrix operand order repaired and checked by reading, not compiled, not run on hardware
//
#include <hip/hip_runtime.h>
#include <math.h>

// ---------------------------------------------------------------------------
// MLLA block forward for gfx1250 (MI455X).  All large GEMMs use
// v_wmma_f32_16x16x32_bf16 with fused epilogues; activations are converted to
// bf16 once to halve GEMM streaming traffic (workload is HBM-bound).
// Wave tile 32x64 (8 WMMA / 12 b128 loads per K-step); elementwise passes
// vectorized 4-wide.
// ---------------------------------------------------------------------------

typedef __attribute__((ext_vector_type(16))) __bf16 v16bf;
typedef __attribute__((ext_vector_type(8)))  float  v8f;

#define DEVI __device__ __forceinline__

constexpr int  Bk  = 8;
constexpr int  Lk  = 2048;
constexpr int  Ck  = 512;
constexpr int  Hh  = 8;
constexpr int  Dd  = 64;
constexpr int  HID = 2048;
constexpr long NTOK = (long)Bk * Lk;      // 16384 token rows
constexpr long SZF  = NTOK * Ck * 4;      // fp32 activation bytes
constexpr long SZH  = NTOK * Ck * 2;      // bf16 activation bytes

// workspace layout (bytes); regions re-used once their chain is finished
constexpr long O_X    = 0;                 // fp32: shortcut x; later x4
constexpr long O_ACT  = O_X    + SZF;      // fp32: act_res (silu gate)
constexpr long O_XNBF = O_ACT  + SZF;      // bf16: ln1(x); later xq_bf; later gated
constexpr long O_X1   = O_XNBF + SZH;      // fp32: in_proj out; later x3
constexpr long O_X2BF = O_X1   + SZF;      // bf16: silu(dwc); later ln2 out
constexpr long O_QBF  = O_X2BF + SZH;      // bf16: q (elu+1)
constexpr long O_KBF  = O_QBF  + SZH;      // bf16: k (elu+1)
constexpr long O_VBF  = O_KBF  + SZH;      // bf16: v
constexpr long O_KVM  = O_VBF  + SZH;      // fp32: 64 * 64x64 kv matrices
constexpr long O_KM   = O_KVM  + 64L*4096*4;  // fp32: 64 * 64 k_mean
constexpr long O_WIN  = O_KM   + 64L*64*4;    // bf16 weights below
constexpr long O_WACT = O_WIN  + 512L*512*2;
constexpr long O_WKV  = O_WACT + 512L*512*2;
constexpr long O_WQ   = O_WKV  + 1024L*512*2;
constexpr long O_WOUT = O_WQ   + 512L*512*2;
constexpr long O_WF1  = O_WOUT + 512L*512*2;
constexpr long O_WF2  = O_WF1  + 2048L*512*2;
constexpr long O_HBF  = O_WF2  + 512L*2048*2; // bf16: gelu(fc1) 67MB

// ------------------------------ math helpers -------------------------------
DEVI float silu_f(float x)  { return x / (1.0f + __expf(-x)); }
DEVI float elu1_f(float x)  { return x > 0.0f ? x + 1.0f : __expf(x); }
DEVI float gelu_f(float x)  { return 0.5f * x * (1.0f + erff(x * 0.70710678118654752f)); }

union F4 { float4 v; float f[4]; };
union H4 { uint2 u; __bf16 h[4]; };
union Pack8 { uint4 u4; __bf16 h[8]; };

// ------------------------- fp32 -> bf16 (4-wide) ---------------------------
__global__ void cvt_bf16_kernel(const float* __restrict__ s, __bf16* __restrict__ d, long n4) {
    long i = (long)blockIdx.x * blockDim.x + threadIdx.x;
    if (i >= n4) return;
    F4 x; x.v = *reinterpret_cast<const float4*>(s + i * 4);
    H4 o;
#pragma unroll
    for (int j = 0; j < 4; ++j) o.h[j] = (__bf16)x.f[j];
    *reinterpret_cast<uint2*>(d + i * 4) = o.u;
}

// ---------------- depthwise conv (k=3, over L) + residual (4-wide) ---------
// out[b,l,c] = x[b,l,c] + bias[c] + sum_t w[c,t] * x[b,l+t-1,c]
__global__ void cpe_kernel(const float* __restrict__ x, const float* __restrict__ w,
                           const float* __restrict__ bias, float* __restrict__ out) {
    long i = (long)blockIdx.x * blockDim.x + threadIdx.x;
    if (i >= (NTOK * Ck) / 4) return;
    long base = i * 4;
    int  c0 = (int)(base % Ck);
    long bl = base / Ck;
    int  l  = (int)(bl % Lk);
    F4 xv; xv.v = *reinterpret_cast<const float4*>(x + base);
    F4 xm, xp;
    if (l > 0)      xm.v = *reinterpret_cast<const float4*>(x + base - Ck);
    else            xm.f[0] = xm.f[1] = xm.f[2] = xm.f[3] = 0.0f;
    if (l < Lk - 1) xp.v = *reinterpret_cast<const float4*>(x + base + Ck);
    else            xp.f[0] = xp.f[1] = xp.f[2] = xp.f[3] = 0.0f;
    F4 o;
#pragma unroll
    for (int j = 0; j < 4; ++j) {
        int c = c0 + j;
        o.f[j] = xv.f[j] + bias[c] + w[c*3+0]*xm.f[j] + w[c*3+1]*xv.f[j] + w[c*3+2]*xp.f[j];
    }
    *reinterpret_cast<float4*>(out + base) = o.v;
}

// ------- in_proj "reshape(B,C,L)" depthwise conv + silu -> bf16 (4-wide) ---
// flat j in [0, L*C): channel c = j / L, pos l = j % L (MLLA reshape trick)
__global__ void dwc_silu_kernel(const float* __restrict__ x, const float* __restrict__ w,
                                const float* __restrict__ bias, __bf16* __restrict__ out) {
    long i = (long)blockIdx.x * blockDim.x + threadIdx.x;
    if (i >= (NTOK * Ck) / 4) return;
    long base = i * 4;
    long b = base / ((long)Lk * Ck);
    long j = base % ((long)Lk * Ck);
    int  c = (int)(j / Lk);
    int  l = (int)(j % Lk);                       // multiple of 4
    const float* xb = x + b * (long)Lk * Ck + (long)c * Lk;
    F4 xv; xv.v = *reinterpret_cast<const float4*>(xb + l);
    float xm1 = (l > 0)       ? xb[l - 1] : 0.0f;
    float xp4 = (l + 4 < Lk)  ? xb[l + 4] : 0.0f;
    float w0 = w[c*3+0], w1 = w[c*3+1], w2 = w[c*3+2], bc = bias[c];
    H4 o;
#pragma unroll
    for (int t = 0; t < 4; ++t) {
        float left  = (t == 0) ? xm1 : xv.f[t - 1];
        float right = (t == 3) ? xp4 : xv.f[t + 1];
        float a = bc + w0 * left + w1 * xv.f[t] + w2 * right;
        o.h[t] = (__bf16)silu_f(a);
    }
    *reinterpret_cast<uint2*>(out + base) = o.u;
}

// ----------------------------- LayerNorm -> bf16 ---------------------------
__global__ __launch_bounds__(256)
void ln_bf16_kernel(const float* __restrict__ x, const float* __restrict__ g,
                    const float* __restrict__ bb, __bf16* __restrict__ out) {
    __shared__ float red[256];
    long row = blockIdx.x;
    const float* xr = x + row * Ck;
    int t = threadIdx.x;
    float x0 = xr[t], x1 = xr[t + 256];
    red[t] = x0 + x1; __syncthreads();
    for (int s = 128; s > 0; s >>= 1) { if (t < s) red[t] += red[t + s]; __syncthreads(); }
    float m = red[0] * (1.0f / Ck); __syncthreads();
    float d0 = x0 - m, d1 = x1 - m;
    red[t] = d0*d0 + d1*d1; __syncthreads();
    for (int s = 128; s > 0; s >>= 1) { if (t < s) red[t] += red[t + s]; __syncthreads(); }
    float inv = rsqrtf(red[0] * (1.0f / Ck) + 1e-5f);
    out[row * Ck + t]       = (__bf16)(d0 * inv * g[t]       + bb[t]);
    out[row * Ck + t + 256] = (__bf16)(d1 * inv * g[t + 256] + bb[t + 256]);
}

// ------------------------- WMMA fragment loaders ---------------------------
// A 16x32 bf16 (ISA 7.12.2): lane m = L&15; K half: lanes<16 -> base 0, else 8
// elems 0..7  = K[klo .. klo+7], elems 8..15 = K[klo+16 .. klo+23]
DEVI v16bf load_a_frag(const __bf16* __restrict__ A, int K, int mbase, int kb, int lane) {
    int m   = mbase + (lane & 15);
    int klo = kb + ((lane & 16) ? 8 : 0);
    const __bf16* p = A + (long)m * K + klo;
    Pack8 x0, x1;
    x0.u4 = *reinterpret_cast<const uint4*>(p);
    x1.u4 = *reinterpret_cast<const uint4*>(p + 16);
    v16bf a;
#pragma unroll
    for (int i = 0; i < 8; ++i) { a[i] = x0.h[i]; a[i + 8] = x1.h[i]; }
    return a;
}

// B 32x16 bf16: lane n = L&15; same K order as A: elems 0..7 = K[klo..klo+7], 8..15 = K[klo+16..klo+23], klo = kb + (lanes<16 ? 0 : 8)
// W is row-major (N,K) so B[k][n] = W[n*K + k] -> contiguous K per lane.
DEVI v16bf load_b_frag(const __bf16* __restrict__ W, int K, int nbase, int kb, int lane) {
    int n  = nbase + (lane & 15);
    int k0 = kb + ((lane & 16) ? 8 : 0);
    const __bf16* p = W + (long)n * K + k0;
    Pack8 x0, x1;
    x0.u4 = *reinterpret_cast<const uint4*>(p);
    x1.u4 = *reinterpret_cast<const uint4*>(p + 16);
    v16bf b;
#pragma unroll
    for (int i = 0; i < 8; ++i) { b[i] = x0.h[i]; b[i + 8] = x1.h[i]; }
    return b;
}

// ----------------------------- WMMA GEMM ----------------------------------
// C = A(MxK,bf16) @ W(N,K,bf16)^T + bias, fused epilogues.
// Block 256 threads = 8 waves arranged 4(M) x 2(N); wave tile 32x64.
// Block tile 128(M) x 128(N).
enum { E_NONE = 0, E_SILU = 1, E_ELU1 = 2, E_KV = 3, E_RES = 4, E_GELU = 5 };

template <int EPI>
__global__ __launch_bounds__(256)
void gemm_wmma_kernel(const __bf16* __restrict__ A, const __bf16* __restrict__ W,
                      const float* __restrict__ bias,
                      float* __restrict__ outF, __bf16* __restrict__ outH,
                      __bf16* __restrict__ outH2, const float* __restrict__ res,
                      int M, int N, int K) {
    const int lane = threadIdx.x & 31;
    const int w    = threadIdx.x >> 5;               // 8 waves
    const int mb   = blockIdx.y * 128 + (w & 3) * 32;
    const int nb   = blockIdx.x * 128 + (w >> 2) * 64;

    v8f acc[2][4] = {{{}, {}, {}, {}}, {{}, {}, {}, {}}};

    for (int kb = 0; kb < K; kb += 32) {
        v16bf a0 = load_a_frag(A, K, mb,      kb, lane);
        v16bf a1 = load_a_frag(A, K, mb + 16, kb, lane);
        v16bf b0 = load_b_frag(W, K, nb,      kb, lane);
        v16bf b1 = load_b_frag(W, K, nb + 16, kb, lane);
        v16bf b2 = load_b_frag(W, K, nb + 32, kb, lane);
        v16bf b3 = load_b_frag(W, K, nb + 48, kb, lane);
        acc[0][0] = __builtin_amdgcn_wmma_f32_16x16x32_bf16(false, a0, false, b0, (short)0, acc[0][0], false, false);
        acc[0][1] = __builtin_amdgcn_wmma_f32_16x16x32_bf16(false, a0, false, b1, (short)0, acc[0][1], false, false);
        acc[0][2] = __builtin_amdgcn_wmma_f32_16x16x32_bf16(false, a0, false, b2, (short)0, acc[0][2], false, false);
        acc[0][3] = __builtin_amdgcn_wmma_f32_16x16x32_bf16(false, a0, false, b3, (short)0, acc[0][3], false, false);
        acc[1][0] = __builtin_amdgcn_wmma_f32_16x16x32_bf16(false, a1, false, b0, (short)0, acc[1][0], false, false);
        acc[1][1] = __builtin_amdgcn_wmma_f32_16x16x32_bf16(false, a1, false, b1, (short)0, acc[1][1], false, false);
        acc[1][2] = __builtin_amdgcn_wmma_f32_16x16x32_bf16(false, a1, false, b2, (short)0, acc[1][2], false, false);
        acc[1][3] = __builtin_amdgcn_wmma_f32_16x16x32_bf16(false, a1, false, b3, (short)0, acc[1][3], false, false);
    }

    // C/D layout: lane -> N = lane&15, vgpr r -> M = r + (lane<16 ? 0 : 8)
#pragma unroll
    for (int ti = 0; ti < 2; ++ti) {
#pragma unroll
        for (int tj = 0; tj < 4; ++tj) {
            int col  = nb + tj * 16 + (lane & 15);
            int row0 = mb + ti * 16 + ((lane & 16) ? 8 : 0);
            float bcol = bias[col];
#pragma unroll
            for (int r = 0; r < 8; ++r) {
                int  row = row0 + r;
                float v = acc[ti][tj][r] + bcol;
                long idx = (long)row * N + col;
                if constexpr (EPI == E_NONE) {
                    outF[idx] = v;
                } else if constexpr (EPI == E_SILU) {
                    outF[idx] = silu_f(v);
                } else if constexpr (EPI == E_ELU1) {
                    outH[idx] = (__bf16)elu1_f(v);
                } else if constexpr (EPI == E_KV) {
                    // N = 2C: cols [0,C) -> k (elu+1), cols [C,2C) -> v
                    if (col < Ck) outH [(long)row * Ck + col]        = (__bf16)elu1_f(v);
                    else          outH2[(long)row * Ck + (col - Ck)] = (__bf16)v;
                } else if constexpr (EPI == E_RES) {
                    outF[idx] = v + res[idx];
                } else if constexpr (EPI == E_GELU) {
                    outH[idx] = (__bf16)gelu_f(v);
                }
            }
        }
    }
}

// ----------------------------- k_mean ------------------------------------
// kmean[b,h,d] = mean_n k[b,n,h*64+d]
__global__ void kmean_kernel(const __bf16* __restrict__ k, float* __restrict__ kmean) {
    int bh = blockIdx.x;                      // 0..63
    int b = bh >> 3, h = bh & 7;
    int d = threadIdx.x;                      // 0..63
    const __bf16* kp = k + ((long)b * Lk) * Ck + h * Dd + d;
    float s = 0.0f;
    for (int n = 0; n < Lk; ++n) s += (float)kp[(long)n * Ck];
    kmean[bh * Dd + d] = s * (1.0f / Lk);
}

// ----------------------------- kv_mat -------------------------------------
// kvm[b,h,d,e] = (1/L) * sum_n k[b,n,h,d] * v[b,n,h,e]   (64x64 per head)
__global__ __launch_bounds__(256)
void kvmat_kernel(const __bf16* __restrict__ k, const __bf16* __restrict__ v,
                  float* __restrict__ kvm) {
    __shared__ __bf16 kt[32][64];
    __shared__ __bf16 vt[32][64];
    int bh = blockIdx.x;
    int b = bh >> 3, h = bh & 7;
    int tid = threadIdx.x;
    int d  = tid >> 2;
    int eb = (tid & 3) << 4;
    const __bf16* kp = k + ((long)b * Lk) * Ck + h * Dd;
    const __bf16* vp = v + ((long)b * Lk) * Ck + h * Dd;
    float acc[16];
#pragma unroll
    for (int i = 0; i < 16; ++i) acc[i] = 0.0f;

    for (int n0 = 0; n0 < Lk; n0 += 32) {
#pragma unroll
        for (int i = 0; i < 8; ++i) {
            int el = tid + i * 256;           // 0..2047
            int rr = el >> 6, cc = el & 63;
            kt[rr][cc] = kp[(long)(n0 + rr) * Ck + cc];
            vt[rr][cc] = vp[(long)(n0 + rr) * Ck + cc];
        }
        __syncthreads();
        for (int nn = 0; nn < 32; ++nn) {
            float kd = (float)kt[nn][d];
#pragma unroll
            for (int i = 0; i < 16; ++i) acc[i] += kd * (float)vt[nn][eb + i];
        }
        __syncthreads();
    }
    float* out = kvm + (long)bh * 4096 + d * 64 + eb;
#pragma unroll
    for (int i = 0; i < 16; ++i) out[i] = acc[i] * (1.0f / Lk);
}

// --------------------- attention out + lepe + gating -----------------------
// gated[b,n,c] = ((q . kvm) * z + lepe(v)) * act_res,  c = h*64+e
__global__ __launch_bounds__(512)
void attn_out_kernel(const __bf16* __restrict__ q, const __bf16* __restrict__ v,
                     const float* __restrict__ kvm, const float* __restrict__ kmean,
                     const float* __restrict__ lw, const float* __restrict__ lb,
                     const float* __restrict__ act, __bf16* __restrict__ gated) {
    __shared__ float qs[512];
    __shared__ float red[512];
    __shared__ float zs[8];
    int b = blockIdx.x / Lk;
    int n = blockIdx.x % Lk;
    int c = threadIdx.x;
    int h = c >> 6, e = c & 63;
    long rowoff = ((long)b * Lk + n) * Ck;

    float qv = (float)q[rowoff + c];
    qs[c] = qv;
    red[c] = qv * kmean[(b * Hh + h) * Dd + e];
    __syncthreads();
    for (int s = 32; s > 0; s >>= 1) { if (e < s) red[c] += red[c + s]; __syncthreads(); }
    if (e == 0) zs[h] = 1.0f / (red[h * 64] + 1e-6f);
    __syncthreads();
    float z = zs[h];

    const float* km = kvm + (long)(b * Hh + h) * 4096;
    float acc = 0.0f;
#pragma unroll 8
    for (int d = 0; d < 64; ++d) acc += qs[h * 64 + d] * km[d * 64 + e];
    acc *= z;

    // lepe: depthwise conv over n of v (B,L,C layout)
    float lp = lb[c] + lw[c*3+1] * (float)v[rowoff + c];
    if (n > 0)      lp += lw[c*3+0] * (float)v[rowoff - Ck + c];
    if (n < Lk - 1) lp += lw[c*3+2] * (float)v[rowoff + Ck + c];

    gated[rowoff + c] = (__bf16)((acc + lp) * act[rowoff + c]);
}

// ---------------------------------------------------------------------------
extern "C" void kernel_launch(void* const* d_in, const int* in_sizes, int n_in,
                              void* d_out, int out_size, void* d_ws, size_t ws_size,
                              hipStream_t stream) {
    (void)in_sizes; (void)n_in; (void)out_size; (void)ws_size;
    const float* x_q  = (const float*)d_in[0];
    const float* x_kv = (const float*)d_in[1];
    const float* cpe1_w = (const float*)d_in[2];
    const float* cpe1_b = (const float*)d_in[3];
    const float* n1_g   = (const float*)d_in[4];
    const float* n1_b   = (const float*)d_in[5];
    const float* inW    = (const float*)d_in[6];
    const float* inB    = (const float*)d_in[7];
    const float* actW   = (const float*)d_in[8];
    const float* actB   = (const float*)d_in[9];
    const float* dwcW   = (const float*)d_in[10];
    const float* dwcB   = (const float*)d_in[11];
    const float* kvW    = (const float*)d_in[12];
    const float* kvB    = (const float*)d_in[13];
    const float* qW     = (const float*)d_in[14];
    const float* qB     = (const float*)d_in[15];
    const float* lepeW  = (const float*)d_in[16];
    const float* lepeB  = (const float*)d_in[17];
    const float* outW   = (const float*)d_in[18];
    const float* outB   = (const float*)d_in[19];
    const float* cpe2_w = (const float*)d_in[20];
    const float* cpe2_b = (const float*)d_in[21];
    const float* n2_g   = (const float*)d_in[22];
    const float* n2_b   = (const float*)d_in[23];
    const float* fc1W   = (const float*)d_in[24];
    const float* fc1B   = (const float*)d_in[25];
    const float* fc2W   = (const float*)d_in[26];
    const float* fc2B   = (const float*)d_in[27];

    char* ws = (char*)d_ws;
    float*  bX    = (float*)(ws + O_X);
    float*  bACT  = (float*)(ws + O_ACT);
    __bf16* bXN   = (__bf16*)(ws + O_XNBF);
    float*  bX1   = (float*)(ws + O_X1);
    __bf16* bX2   = (__bf16*)(ws + O_X2BF);
    __bf16* bQ    = (__bf16*)(ws + O_QBF);
    __bf16* bK    = (__bf16*)(ws + O_KBF);
    __bf16* bV    = (__bf16*)(ws + O_VBF);
    float*  bKVM  = (float*)(ws + O_KVM);
    float*  bKM   = (float*)(ws + O_KM);
    __bf16* wIN   = (__bf16*)(ws + O_WIN);
    __bf16* wACT  = (__bf16*)(ws + O_WACT);
    __bf16* wKV   = (__bf16*)(ws + O_WKV);
    __bf16* wQ    = (__bf16*)(ws + O_WQ);
    __bf16* wOUT  = (__bf16*)(ws + O_WOUT);
    __bf16* wF1   = (__bf16*)(ws + O_WF1);
    __bf16* wF2   = (__bf16*)(ws + O_WF2);
    __bf16* bH    = (__bf16*)(ws + O_HBF);

    const long NELT = NTOK * Ck;                      // 8.39M
    dim3 blk(256);
    dim3 grdE4((unsigned)((NELT / 4 + 255) / 256));
    auto g4 = [](long n) { return dim3((unsigned)((n / 4 + 255) / 256)); };

    // weights -> bf16 (4-wide)
    cvt_bf16_kernel<<<g4(512L*512),  blk, 0, stream>>>(inW,  wIN,  512L*512/4);
    cvt_bf16_kernel<<<g4(512L*512),  blk, 0, stream>>>(actW, wACT, 512L*512/4);
    cvt_bf16_kernel<<<g4(1024L*512), blk, 0, stream>>>(kvW,  wKV,  1024L*512/4);
    cvt_bf16_kernel<<<g4(512L*512),  blk, 0, stream>>>(qW,   wQ,   512L*512/4);
    cvt_bf16_kernel<<<g4(512L*512),  blk, 0, stream>>>(outW, wOUT, 512L*512/4);
    cvt_bf16_kernel<<<g4(2048L*512), blk, 0, stream>>>(fc1W, wF1,  2048L*512/4);
    cvt_bf16_kernel<<<g4(512L*2048), blk, 0, stream>>>(fc2W, wF2,  512L*2048/4);

    // x = x_kv + cpe1(x_kv); shortcut in bX
    cpe_kernel<<<grdE4, blk, 0, stream>>>(x_kv, cpe1_w, cpe1_b, bX);
    // ln1 -> bf16
    ln_bf16_kernel<<<dim3((unsigned)NTOK), blk, 0, stream>>>(bX, n1_g, n1_b, bXN);
    // act_res = silu(xn @ actW^T + b)
    gemm_wmma_kernel<E_SILU><<<dim3(512/128, NTOK/128), blk, 0, stream>>>(
        bXN, wACT, actB, bACT, nullptr, nullptr, nullptr, (int)NTOK, 512, 512);
    // x1 = xn @ inW^T + b
    gemm_wmma_kernel<E_NONE><<<dim3(512/128, NTOK/128), blk, 0, stream>>>(
        bXN, wIN, inB, bX1, nullptr, nullptr, nullptr, (int)NTOK, 512, 512);
    // x2 = silu(dwc(x1 viewed as (B,C,L))) -> bf16
    dwc_silu_kernel<<<grdE4, blk, 0, stream>>>(bX1, dwcW, dwcB, bX2);
    // xn buffer dead -> reuse for bf16 x_q
    cvt_bf16_kernel<<<grdE4, blk, 0, stream>>>(x_q, bXN, NELT / 4);
    // kv = x2 @ kvW^T + b : split into k (elu+1) and v
    gemm_wmma_kernel<E_KV><<<dim3(1024/128, NTOK/128), blk, 0, stream>>>(
        bX2, wKV, kvB, nullptr, bK, bV, nullptr, (int)NTOK, 1024, 512);
    // q = elu(x_q @ qW^T + b) + 1
    gemm_wmma_kernel<E_ELU1><<<dim3(512/128, NTOK/128), blk, 0, stream>>>(
        bXN, wQ, qB, nullptr, bQ, nullptr, nullptr, (int)NTOK, 512, 512);
    // attention core
    kmean_kernel<<<dim3(64), dim3(64), 0, stream>>>(bK, bKM);
    kvmat_kernel<<<dim3(64), blk, 0, stream>>>(bK, bV, bKVM);
    attn_out_kernel<<<dim3((unsigned)(Bk * Lk)), dim3(512), 0, stream>>>(
        bQ, bV, bKVM, bKM, lepeW, lepeB, bACT, bXN);   // gated reuses bXN
    // x3 = shortcut + gated @ outW^T + b   (reuses bX1)
    gemm_wmma_kernel<E_RES><<<dim3(512/128, NTOK/128), blk, 0, stream>>>(
        bXN, wOUT, outB, bX1, nullptr, nullptr, bX, (int)NTOK, 512, 512);
    // x4 = x3 + cpe2(x3)  (reuses bX)
    cpe_kernel<<<grdE4, blk, 0, stream>>>(bX1, cpe2_w, cpe2_b, bX);
    // ln2 -> bf16 (reuses bX2)
    ln_bf16_kernel<<<dim3((unsigned)NTOK), blk, 0, stream>>>(bX, n2_g, n2_b, bX2);
    // h = gelu(y @ fc1W^T + b) -> bf16
    gemm_wmma_kernel<E_GELU><<<dim3(2048/128, NTOK/128), blk, 0, stream>>>(
        bX2, wF1, fc1B, nullptr, bH, nullptr, nullptr, (int)NTOK, 2048, 512);
    // out = x4 + h @ fc2W^T + b
    gemm_wmma_kernel<E_RES><<<dim3(512/128, NTOK/128), blk, 0, stream>>>(
        bH, wF2, fc2B, (float*)d_out, nullptr, nullptr, bX, (int)NTOK, 512, 2048);
}
